// DisenGCN_52458730553680
// MI455X (gfx1250) — compile-verified, matrix operand order checked by reading, not run on hardware
//
#include <hip/hip_runtime.h>
#include <hip/hip_bf16.h>

typedef __attribute__((ext_vector_type(16))) _Float16 v16h;
typedef __attribute__((ext_vector_type(8)))  float    v8f;

#define N_USERS 40000
#define N_ITEMS 60000
#define N_NODES (N_USERS + N_ITEMS)   // 100000
#define HID     256
#define KCH     8
#define DCH     32
#define M_EDGES 500000
#define BATCH   4096
#define NUM_LAYERS 5
#define ROUTIT  7

// ---------------------------------------------------------------------------
// wave32 xor-swizzle (group-of-32 mode: and=0x1f, or=0, xor=XMASK)
// ---------------------------------------------------------------------------
template <int XMASK>
__device__ inline float swz_xor(float v) {
    int r = __builtin_amdgcn_ds_swizzle(__float_as_int(v), (XMASK << 10) | 0x1f);
    return __int_as_float(r);
}

// ---------------------------------------------------------------------------
// Kernel 1: Z = relu(X @ W + b)  via v_wmma_f32_16x16x32_f16
// one wave per 16x16 output tile; 100000/16 = 6250 row tiles, 256/16 = 16 col tiles
// grid = 12500 blocks * 256 threads (8 waves). A loaded as float4 (32B aligned).
// ---------------------------------------------------------------------------
__global__ __launch_bounds__(256) void gemm_relu_wmma(
    const float* __restrict__ X, const float* __restrict__ W,
    const float* __restrict__ bias, float* __restrict__ Z)
{
    const int lane = threadIdx.x & 31;
    const int wave = threadIdx.x >> 5;
    const int tile = blockIdx.x * 8 + wave;   // 0..99999
    const int row0 = (tile >> 4) * 16;
    const int col0 = (tile & 15) * 16;

    const int r  = lane & 15;     // row (A) / col (B,C)
    const int hi = lane >> 4;     // half-wave selector
    const int kb = hi * 8;        // K sub-base per ISA A/B layout

    v8f acc = {};
    for (int kk = 0; kk < HID; kk += 32) {
        v16h a, bm;
        const float* xrow = X + (size_t)(row0 + r) * HID + kk + kb;
        const float4 a0 = *(const float4*)(xrow);        // K = kb+0..3
        const float4 a1 = *(const float4*)(xrow + 4);    // K = kb+4..7
        const float4 a2 = *(const float4*)(xrow + 16);   // K = kb+16..19
        const float4 a3 = *(const float4*)(xrow + 20);   // K = kb+20..23
        a[0]  = (_Float16)a0.x; a[1]  = (_Float16)a0.y; a[2]  = (_Float16)a0.z; a[3]  = (_Float16)a0.w;
        a[4]  = (_Float16)a1.x; a[5]  = (_Float16)a1.y; a[6]  = (_Float16)a1.z; a[7]  = (_Float16)a1.w;
        a[8]  = (_Float16)a2.x; a[9]  = (_Float16)a2.y; a[10] = (_Float16)a2.z; a[11] = (_Float16)a2.w;
        a[12] = (_Float16)a3.x; a[13] = (_Float16)a3.y; a[14] = (_Float16)a3.z; a[15] = (_Float16)a3.w;

        const float* wcol = W + (size_t)(kk + kb) * HID + col0 + r;
#pragma unroll
        for (int j = 0; j < 8; ++j) {
            bm[j]     = (_Float16)wcol[(size_t)j * HID];
            bm[8 + j] = (_Float16)wcol[(size_t)(16 + j) * HID];
        }
        acc = __builtin_amdgcn_wmma_f32_16x16x32_f16(
            /*neg_a=*/false, a, /*neg_b=*/false, bm,
            /*c_mod=*/(short)0, acc, /*reuse_a=*/false, /*reuse_b=*/false);
    }
    const float bv = bias[col0 + r];
#pragma unroll
    for (int v = 0; v < 8; ++v) {
        const int rowM = v + hi * 8;           // C/D layout: VGPR v -> M = v (+8)
        float val = acc[v] + bv;
        val = val > 0.f ? val : 0.f;
        Z[(size_t)(row0 + rowM) * HID + col0 + r] = val;
    }
}

// ---------------------------------------------------------------------------
// Kernel 2: in-place per-channel L2 normalization (one thread per (node,channel))
// ---------------------------------------------------------------------------
__global__ __launch_bounds__(256) void l2norm_inplace(float* __restrict__ buf, int nk)
{
    const int i = blockIdx.x * 256 + threadIdx.x;
    if (i >= nk) return;
    float* p = buf + (size_t)i * DCH;
    float4 v[8];
    float ss = 0.f;
#pragma unroll
    for (int j = 0; j < 8; ++j) {
        v[j] = *(const float4*)(p + j * 4);
        ss += v[j].x * v[j].x + v[j].y * v[j].y + v[j].z * v[j].z + v[j].w * v[j].w;
    }
    const float inv = 1.f / fmaxf(sqrtf(ss), 1e-12f);
#pragma unroll
    for (int j = 0; j < 8; ++j) {
        v[j].x *= inv; v[j].y *= inv; v[j].z *= inv; v[j].w *= inv;
        *(float4*)(p + j * 4) = v[j];
    }
}

// ---------------------------------------------------------------------------
// Kernel 3: zero-fill (float4), exact size
// ---------------------------------------------------------------------------
__global__ __launch_bounds__(256) void zero4(float4* __restrict__ p)
{
    p[(size_t)blockIdx.x * 256 + threadIdx.x] = make_float4(0.f, 0.f, 0.f, 0.f);
}

// ---------------------------------------------------------------------------
// Kernel 4: per-edge routing step. One wave per edge.
//   lane -> channel k = lane>>2, dims d0 = (lane&3)*8 (8 floats per lane)
//   dot over D=32 via group-of-4 xor reduction; softmax over K=8 via
//   wave-wide butterfly (each channel replicated 4x; sum scaled by 1/4).
//   scatter: acc[src] += 4 * softmax_p * z[trg]   (f32 atomics)
// ---------------------------------------------------------------------------
__global__ __launch_bounds__(256) void route_edge(
    const float* __restrict__ c, const float* __restrict__ z,
    const int* __restrict__ src, const int* __restrict__ trg,
    float* __restrict__ acc, int m_edges)
{
    const int lane = threadIdx.x & 31;
    const int wave = threadIdx.x >> 5;
    const int m = blockIdx.x * 8 + wave;
    if (m >= m_edges) return;           // whole-wave uniform (EXEC stays all-1s)

    const int s = src[m];
    const int t = trg[m];
    const int k  = lane >> 2;
    const int d0 = (lane & 3) * 8;
    const size_t zo = (size_t)t * HID + k * DCH + d0;
    const size_t co = (size_t)s * HID + k * DCH + d0;

    const float4 zt0 = *(const float4*)(z + zo);
    const float4 zt1 = *(const float4*)(z + zo + 4);
    const float4 cs0 = *(const float4*)(c + co);
    const float4 cs1 = *(const float4*)(c + co + 4);

    float p = zt0.x * cs0.x + zt0.y * cs0.y + zt0.z * cs0.z + zt0.w * cs0.w
            + zt1.x * cs1.x + zt1.y * cs1.y + zt1.z * cs1.z + zt1.w * cs1.w;
    // reduce within group of 4 lanes -> p[k] replicated
    p += swz_xor<1>(p);
    p += swz_xor<2>(p);
    // softmax over K=8 (TAU = 1): wave-wide butterfly
    float mx = p;
    mx = fmaxf(mx, swz_xor<4>(mx));
    mx = fmaxf(mx, swz_xor<8>(mx));
    mx = fmaxf(mx, swz_xor<16>(mx));
    const float e = __expf(p - mx);
    float se = e;
    se += swz_xor<4>(se);
    se += swz_xor<8>(se);
    se += swz_xor<16>(se);
    se *= 0.25f;                        // butterfly sums each channel once, so wgt = 4x softmax
    const float wgt = e / se;

    float* ap = acc + co;
    atomicAdd(ap + 0, wgt * zt0.x);
    atomicAdd(ap + 1, wgt * zt0.y);
    atomicAdd(ap + 2, wgt * zt0.z);
    atomicAdd(ap + 3, wgt * zt0.w);
    atomicAdd(ap + 4, wgt * zt1.x);
    atomicAdd(ap + 5, wgt * zt1.y);
    atomicAdd(ap + 6, wgt * zt1.z);
    atomicAdd(ap + 7, wgt * zt1.w);
}

// ---------------------------------------------------------------------------
// Kernel 5: out = [relu?] l2norm(z + acc)  (one thread per (node,channel)).
//   For routing iterations t<ROUTIT-1: out == acc (in place), do_relu = 0.
//   For the final iteration:           out == Z   (in place vs z), do_relu = 1,
//   fusing the layer-end Z = relu(c) pass. Per-thread read-then-write of the
//   same 32 floats makes the in-place update safe.
// ---------------------------------------------------------------------------
__global__ __launch_bounds__(256) void combine_norm(
    const float* __restrict__ z, const float* __restrict__ acc,
    float* __restrict__ out, int do_relu, int nk)
{
    const int i = blockIdx.x * 256 + threadIdx.x;
    if (i >= nk) return;
    const size_t base = (size_t)i * DCH;
    float4 v[8];
    float ss = 0.f;
#pragma unroll
    for (int j = 0; j < 8; ++j) {
        float4 a  = *(const float4*)(acc + base + j * 4);
        float4 zz = *(const float4*)(z + base + j * 4);
        a.x += zz.x; a.y += zz.y; a.z += zz.z; a.w += zz.w;
        ss += a.x * a.x + a.y * a.y + a.z * a.z + a.w * a.w;
        v[j] = a;
    }
    const float inv = 1.f / fmaxf(sqrtf(ss), 1e-12f);
#pragma unroll
    for (int j = 0; j < 8; ++j) {
        v[j].x *= inv; v[j].y *= inv; v[j].z *= inv; v[j].w *= inv;
        if (do_relu) {
            v[j].x = fmaxf(v[j].x, 0.f); v[j].y = fmaxf(v[j].y, 0.f);
            v[j].z = fmaxf(v[j].z, 0.f); v[j].w = fmaxf(v[j].w, 0.f);
        }
        *(float4*)(out + base + j * 4) = v[j];
    }
}

// ---------------------------------------------------------------------------
// Kernel 6: gather user/pos/neg rows into d_out (3 * B * 256 elements)
// ---------------------------------------------------------------------------
__global__ __launch_bounds__(256) void gather_out(
    const float* __restrict__ Z, const int* __restrict__ users,
    const int* __restrict__ pos, const int* __restrict__ neg,
    float* __restrict__ out)
{
    const int i = blockIdx.x * 256 + threadIdx.x;      // < 3*B*256
    const int seg  = i / (BATCH * HID);
    const int rem  = i - seg * (BATCH * HID);
    const int row  = rem / HID;
    const int col  = rem - row * HID;
    int node;
    if (seg == 0)      node = users[row];
    else if (seg == 1) node = N_USERS + pos[row];
    else               node = N_USERS + neg[row];
    out[i] = Z[(size_t)node * HID + col];
}

// ---------------------------------------------------------------------------
extern "C" void kernel_launch(void* const* d_in, const int* in_sizes, int n_in,
                              void* d_out, int out_size, void* d_ws, size_t ws_size,
                              hipStream_t stream) {
    const float* X     = (const float*)d_in[0];   // N x 256
    const float* W     = (const float*)d_in[1];   // 256 x 256
    const float* bias  = (const float*)d_in[2];   // 256
    const int*   edges = (const int*)d_in[3];     // [2, M]: src then trg
    const int*   users = (const int*)d_in[4];
    const int*   pos   = (const int*)d_in[5];
    const int*   neg   = (const int*)d_in[6];
    float*       out   = (float*)d_out;

    const int* src = edges;
    const int* trg = edges + M_EDGES;

    const size_t NKF = (size_t)N_NODES * HID;     // 25.6M floats per buffer
    float* Z  = (float*)d_ws;
    float* CA = Z + NKF;
    float* CB = CA + NKF;

    const int NK = N_NODES * KCH;                 // 800000 channels

    // 1) Z = relu(X @ W + b)   (WMMA), then per-channel L2 norm
    gemm_relu_wmma<<<(N_NODES / 16) * (HID / 16) / 8, 256, 0, stream>>>(X, W, bias, Z);
    l2norm_inplace<<<NK / 256, 256, 0, stream>>>(Z, NK);

    // 2) 5 layers x 7 routing iterations
    for (int layer = 0; layer < NUM_LAYERS; ++layer) {
        const float* c = Z;                       // c starts as z
        float* accbuf[2] = {CA, CB};
        for (int t = 0; t < ROUTIT; ++t) {
            float* acc = accbuf[t & 1];
            const int last = (t == ROUTIT - 1);
            zero4<<<(int)(NKF / 4 / 256), 256, 0, stream>>>((float4*)acc);
            route_edge<<<M_EDGES / 8, 256, 0, stream>>>(c, Z, src, trg, acc, M_EDGES);
            if (last) {
                // fused: Z = relu(l2norm(Z + acc))   (layer output)
                combine_norm<<<NK / 256, 256, 0, stream>>>(Z, acc, Z, 1, NK);
            } else {
                combine_norm<<<NK / 256, 256, 0, stream>>>(Z, acc, acc, 0, NK);
                c = acc;
            }
        }
    }

    // 3) gather outputs
    gather_out<<<(3 * BATCH * HID) / 256, 256, 0, stream>>>(Z, users, pos, neg, out);
}
